// Money_former_MLA_DINT_cog_attn_MTP_66056597012695
// MI455X (gfx1250) — matrix operand order repaired and checked by reading, not compiled, not run on hardware
//
#include <hip/hip_runtime.h>

// ---------------- constants ----------------
static constexpr int S    = 2048;
static constexpr int D    = 1024;
static constexpr int NH   = 16;
static constexpr int HD   = 64;
static constexpr int KVC  = 256;
static constexpr int QC   = 384;
static constexpr float EPS = 1e-5f;
static constexpr float LAMBDA_INIT = 0.2f;              // 0.8 - 0.6*exp(0)
static constexpr float QK_SCALE = 0.14433756729740643f; // 1/sqrt(48)

// ---------------- bf16 helpers ----------------
__device__ __forceinline__ unsigned short f2bf(float f) {
  unsigned int u = __float_as_uint(f);
  unsigned int r = u + 0x7fffu + ((u >> 16) & 1u);
  return (unsigned short)(r >> 16);
}
__device__ __forceinline__ float bf2f(unsigned short h) {
  return __uint_as_float(((unsigned int)h) << 16);
}

// ---------------- WMMA types ----------------
typedef __attribute__((ext_vector_type(16))) __bf16 bf16x16;
typedef __attribute__((ext_vector_type(8)))  float  floatx8;
union Frag16 { bf16x16 v; uint4 q[2]; unsigned short u[16]; };

// ---------------- f32 -> bf16 convert (row-major copy) ----------------
__global__ void cvt_bf16_kernel(const float* __restrict__ in,
                                unsigned short* __restrict__ out, long long n) {
  long long i = (long long)blockIdx.x * blockDim.x + threadIdx.x;
  if (i < n) out[i] = f2bf(in[i]);
}

// ---------------- tiled f32 -> bf16 transpose: in(RxC, pitch ldin) -> out(CxR) ----------------
__global__ void cvt_transpose_kernel(const float* __restrict__ in, long long ldin,
                                     unsigned short* __restrict__ out, int R, int C) {
  __shared__ float tile[32][33];
  int c0 = blockIdx.x * 32, r0 = blockIdx.y * 32;
  int tx = threadIdx.x, ty = threadIdx.y;  // 32 x 8
  for (int i = 0; i < 32; i += 8) {
    int r = r0 + ty + i, cc = c0 + tx;
    tile[ty + i][tx] = (r < R && cc < C) ? in[(long long)r * ldin + cc] : 0.f;
  }
  __syncthreads();
  for (int i = 0; i < 32; i += 8) {
    int cc = c0 + ty + i, r = r0 + tx;
    if (cc < C && r < R) out[(long long)cc * R + r] = f2bf(tile[tx][ty + i]);
  }
}

// ---------------- generic batched bf16 WMMA GEMM ----------------
// C(f32, MxN) = A(bf16, MxK row-major) * Bt(bf16, NxK row-major, i.e. B transposed)
// causalMode: 0 none, 1 skip tiles fully above diagonal (QK^T),
//             2 clamp K-loop to block row extent (A*V, upper triangle of A is 0)
#define BM 128
#define BN 64
#define BK 32
#define KP 40   // LDS row pitch in bf16 elems (80B, 16B-aligned, conflict-free)

__global__ __launch_bounds__(256)
void gemm_bf16_kernel(const unsigned short* __restrict__ A, int lda, long long strideA,
                      const unsigned short* __restrict__ Bt, int ldbt, long long strideB,
                      float* __restrict__ C, int ldc, long long strideC,
                      int M, int N, int K, int causalMode) {
  __shared__ __align__(16) unsigned short As[BM * KP];
  __shared__ __align__(16) unsigned short Bs[BN * KP];

  int batch = blockIdx.z;
  A  += (long long)batch * strideA;
  Bt += (long long)batch * strideB;
  C  += (long long)batch * strideC;

  int row0 = blockIdx.y * BM;
  int col0 = blockIdx.x * BN;
  if (causalMode == 1 && col0 > row0 + BM - 1) return;  // fully masked tile
  int kEnd = K;
  if (causalMode == 2) { int lim = row0 + BM; kEnd = (lim < K) ? lim : K; }

  int tid  = threadIdx.x;
  int lane = tid & 31;
  int wid  = tid >> 5;
  int wm   = wid >> 1;   // 0..3
  int wn   = wid & 1;    // 0..1
  int halfl = lane >> 4; // 0/1
  int mrow  = lane & 15;

  // staging coordinates (8 bf16 = 16B chunks)
  int ar = tid >> 2;            // A rows: ar and ar+64
  int acs = (tid & 3) * 8;      // A k-chunk
  int br = tid >> 2;            // B row (n), 0..63
  int bcs = (tid & 3) * 8;      // B k-chunk

  floatx8 acc[2][2];
  for (int a = 0; a < 2; a++)
    for (int b = 0; b < 2; b++)
      for (int e = 0; e < 8; e++) acc[a][b][e] = 0.0f;

  for (int k0 = 0; k0 < kEnd; k0 += BK) {
    // ---- stage A tile (128x32): async global->LDS, 2 chunks/thread ----
    for (int hchunk = 0; hchunk < 2; hchunk++) {
      int r = ar + hchunk * 64;
      int gr = row0 + r;
      if (gr < M && (k0 + acs + 7) < K) {
        unsigned ldsa = (unsigned)(size_t)&As[r * KP + acs];
        unsigned long long ga =
            (unsigned long long)(size_t)(A + (long long)gr * lda + k0 + acs);
        asm volatile("global_load_async_to_lds_b128 %0, %1, off"
                     :: "v"(ldsa), "v"(ga) : "memory");
      } else {
        *(uint4*)&As[r * KP + acs] = make_uint4(0u, 0u, 0u, 0u);
      }
    }
    // ---- stage Bt tile (64x32): async global->LDS, 1 chunk/thread ----
    {
      int gn = col0 + br;
      if (gn < N && (k0 + bcs + 7) < K) {
        unsigned ldsb = (unsigned)(size_t)&Bs[br * KP + bcs];
        unsigned long long gb =
            (unsigned long long)(size_t)(Bt + (long long)gn * ldbt + k0 + bcs);
        asm volatile("global_load_async_to_lds_b128 %0, %1, off"
                     :: "v"(ldsb), "v"(gb) : "memory");
      } else {
        *(uint4*)&Bs[br * KP + bcs] = make_uint4(0u, 0u, 0u, 0u);
      }
    }
    // ---- prefetch next K tile while async copies land ----
    if (k0 + BK < kEnd) {
      __builtin_prefetch(A + (long long)(row0 + ar) * lda + k0 + BK + acs, 0, 1);
      if (col0 + br < N)
        __builtin_prefetch(Bt + (long long)(col0 + br) * ldbt + k0 + BK + bcs, 0, 1);
    }
    asm volatile("s_wait_asynccnt 0x0" ::: "memory");
    __syncthreads();

    // ---- fragments: pure b128 LDS loads (K-contiguous for both A and B) ----
    Frag16 af[2], bf[2];
    for (int ms = 0; ms < 2; ms++) {
      const unsigned short* base = &As[(wm * 32 + ms * 16 + mrow) * KP + halfl * 8];
      af[ms].q[0] = *(const uint4*)base;          // K = 8*half .. +7
      af[ms].q[1] = *(const uint4*)(base + 16);   // K = 16+8*half .. +7
    }
    for (int ns = 0; ns < 2; ns++) {
      const unsigned short* base = &Bs[(wn * 32 + ns * 16 + mrow) * KP + halfl * 8];
      bf[ns].q[0] = *(const uint4*)base;          // K = 8*half .. +7
      bf[ns].q[1] = *(const uint4*)(base + 16);   // K = 16+8*half .. +7
    }
    for (int ms = 0; ms < 2; ms++)
      for (int ns = 0; ns < 2; ns++)
        acc[ms][ns] = __builtin_amdgcn_wmma_f32_16x16x32_bf16(
            false, af[ms].v, false, bf[ns].v,
            (short)0, acc[ms][ns], false, false);
    __syncthreads();
  }

  // store (C/D layout: VGPR e -> M = e + 8*half, N = lane&15)
  for (int ms = 0; ms < 2; ms++)
    for (int ns = 0; ns < 2; ns++) {
      int ncol = col0 + wn * 32 + ns * 16 + mrow;
      if (ncol < N) {
        for (int e = 0; e < 8; e++) {
          int mr = row0 + wm * 32 + ms * 16 + e + halfl * 8;
          if (mr < M) C[(long long)mr * ldc + ncol] = acc[ms][ns][e];
        }
      }
    }
}

// ---------------- generic RMSNorm (f32 in, bf16 out) ----------------
__global__ void rmsnorm_kernel(const float* __restrict__ in, int ldin,
                               const float* __restrict__ w,
                               unsigned short* __restrict__ out, int ldout, int ncols) {
  long long row = blockIdx.x;
  const float* r = in + row * ldin;
  unsigned short* o = out + row * ldout;
  __shared__ float red[256];
  float s = 0.f;
  for (int k = threadIdx.x; k < ncols; k += blockDim.x) { float x = r[k]; s += x * x; }
  red[threadIdx.x] = s; __syncthreads();
  for (int st = blockDim.x >> 1; st > 0; st >>= 1) {
    if ((int)threadIdx.x < st) red[threadIdx.x] += red[threadIdx.x + st];
    __syncthreads();
  }
  float inv = rsqrtf(red[0] / (float)ncols + EPS);
  for (int k = threadIdx.x; k < ncols; k += blockDim.x) o[k] = f2bf(r[k] * inv * w[k]);
}

// ---------------- build q_f (S-major) and k_f (S-major) with grouped RoPE ----------------
__global__ void build_qk_kernel(const float* __restrict__ qp, const float* __restrict__ kv,
                                const float* __restrict__ c, const float* __restrict__ freqs,
                                unsigned short* __restrict__ qf, unsigned short* __restrict__ kf) {
  int t = blockIdx.x * blockDim.x + threadIdx.x;  // 65536 threads
  int s = t >> 5;
  int hh = t & 31;
  int nh = hh >> 1, par = hh & 1;

  unsigned short* qrow = qf + ((long long)hh * S + s) * 64;
  unsigned short* krow = kf + ((long long)hh * S + s) * 64;
  const float* qps = qp + (long long)s * 1536 + nh * 96;
  const float* kvs = kv + (long long)s * 2048 + nh * 128;

  for (int j = 0; j < 32; j++) qrow[j] = f2bf(qps[par * 32 + j]);
  for (int j = 0; j < 32; j++) krow[j] = f2bf(kvs[par * 32 + j]);

  float qr[16], kr[16];
  for (int j = 0; j < 16; j++) {
    qr[j] = qps[64 + par * 16 + j];
    kr[j] = c[(long long)s * 288 + 256 + par * 16 + j];
  }
  int p = s >> 2;                         // s = sps*4 + nseq -> RoPE position = sps
  if (p > 0) {
    const float* fr = freqs + (long long)(p - 1) * 8;
    for (int tt = 0; tt < 8; tt++) {
      float cs = __cosf(fr[tt]), sn = __sinf(fr[tt]);
      float a = qr[2 * tt], b = qr[2 * tt + 1];
      qr[2 * tt] = a * cs - b * sn; qr[2 * tt + 1] = a * sn + b * cs;
      a = kr[2 * tt]; b = kr[2 * tt + 1];
      kr[2 * tt] = a * cs - b * sn; kr[2 * tt + 1] = a * sn + b * cs;
    }
  }
  for (int j = 0; j < 16; j++) {
    qrow[32 + j] = f2bf(qr[j]); qrow[48 + j] = 0;
    krow[32 + j] = f2bf(kr[j]); krow[48 + j] = 0;
  }
}

// ---------------- lambda scalar ----------------
__global__ void lambda_kernel(const float* lk1, const float* lk2,
                              const float* lq1, const float* lq2, float* lam) {
  if (threadIdx.x == 0 && blockIdx.x == 0) {
    float s1 = 0.f, s2 = 0.f;
    for (int i = 0; i < 32; i++) { s1 += lk1[i] * lq1[i]; s2 += lk2[i] * lq2[i]; }
    *lam = __expf(s1) - __expf(s2) + LAMBDA_INIT;
  }
}

// ---------------- causal sign-softmax over |a| ----------------
__global__ void softmax_sign_kernel(const float* __restrict__ L,
                                    unsigned short* __restrict__ A, float scale) {
  int head = blockIdx.y;
  int q = blockIdx.x;
  const float* Lr = L + (long long)head * S * S + (long long)q * S;
  unsigned short* Ar = A + (long long)head * S * S + (long long)q * S;
  int tid = threadIdx.x;
  __shared__ float red[256];
  int len = q + 1;
  float m = -1e30f;
  for (int k = tid; k < len; k += 256) m = fmaxf(m, fabsf(Lr[k] * scale));
  red[tid] = m; __syncthreads();
  for (int st = 128; st > 0; st >>= 1) { if (tid < st) red[tid] = fmaxf(red[tid], red[tid + st]); __syncthreads(); }
  m = red[0]; __syncthreads();
  float sum = 0.f;
  for (int k = tid; k < len; k += 256) sum += __expf(fabsf(Lr[k] * scale) - m);
  red[tid] = sum; __syncthreads();
  for (int st = 128; st > 0; st >>= 1) { if (tid < st) red[tid] += red[tid + st]; __syncthreads(); }
  float inv = 1.f / red[0];
  for (int k = tid; k < S; k += 256) {
    float v = 0.f;
    if (k < len) {
      float a = Lr[k] * scale;
      float pz = __expf(fabsf(a) - m) * inv;
      v = (a > 0.f) ? pz : ((a < 0.f) ? -pz : 0.f);
    }
    Ar[k] = f2bf(v);
  }
}

// ---------------- column mean g[k] = mean_q a0[q,k], two stage ----------------
__global__ void colmean_part_kernel(const unsigned short* __restrict__ A0,
                                    float* __restrict__ gpart) {
  int k = blockIdx.x * blockDim.x + threadIdx.x;
  int chunk = blockIdx.y;                 // 16 chunks of 128 rows
  float s = 0.f;
  int q0 = chunk * 128;
  for (int q = q0; q < q0 + 128; q++) s += bf2f(A0[(long long)q * S + k]);
  gpart[(long long)chunk * S + k] = s;
}
__global__ void colmean_final_kernel(const float* __restrict__ gpart, float* __restrict__ g) {
  int k = blockIdx.x * blockDim.x + threadIdx.x;
  float s = 0.f;
  for (int ch = 0; ch < 16; ch++) s += gpart[(long long)ch * S + k];
  g[k] = s / (float)S;
}

// ---------------- differential combine ----------------
__global__ void combine_kernel(const unsigned short* __restrict__ A0,
                               const unsigned short* __restrict__ A1,
                               const float* __restrict__ g, const float* __restrict__ lamp,
                               unsigned short* __restrict__ AF) {
  int q = blockIdx.x;
  float lam = *lamp;
  const unsigned short* r0 = A0 + (long long)q * S;
  const unsigned short* r1 = A1 + (long long)q * S;
  unsigned short* ro = AF + (long long)q * S;
  for (int k = threadIdx.x; k < S; k += blockDim.x) {
    float v = 0.f;
    if (k <= q) v = bf2f(r0[k]) - lam * bf2f(r1[k]) + lam * g[k];
    ro[k] = f2bf(v);
  }
}

// ---------------- host launcher ----------------
static inline void launch_gemm(const unsigned short* A, int lda, long long sA,
                               const unsigned short* Bt, int ldbt, long long sB,
                               float* C, int ldc, long long sC,
                               int M, int N, int K, int causal, int batch, hipStream_t st) {
  dim3 grid((N + BN - 1) / BN, (M + BM - 1) / BM, batch);
  gemm_bf16_kernel<<<grid, 256, 0, st>>>(A, lda, sA, Bt, ldbt, sB, C, ldc, sC, M, N, K, causal);
}

extern "C" void kernel_launch(void* const* d_in, const int* in_sizes, int n_in,
                              void* d_out, int out_size, void* d_ws, size_t ws_size,
                              hipStream_t stream) {
  (void)in_sizes; (void)n_in; (void)out_size; (void)ws_size;
  const float* x     = (const float*)d_in[0];
  const float* freqs = (const float*)d_in[2];
  const float* Wkvd  = (const float*)d_in[3];
  const float* Wqd   = (const float*)d_in[4];
  const float* kvw   = (const float*)d_in[5];
  const float* qw    = (const float*)d_in[6];
  const float* Wkvup = (const float*)d_in[7];
  const float* Wqup  = (const float*)d_in[8];
  const float* lk1   = (const float*)d_in[9];
  const float* lk2   = (const float*)d_in[10];
  const float* lq1   = (const float*)d_in[11];
  const float* lq2   = (const float*)d_in[12];
  const float* normw = (const float*)d_in[13];
  const float* Wo    = (const float*)d_in[14];
  float* out = (float*)d_out;

  char* ws = (char*)d_ws;
  size_t off = 0;
  auto alloc = [&](size_t bytes) -> char* {
    char* p = ws + off;
    off = (off + bytes + 255) & ~(size_t)255;
    return p;
  };
  unsigned short* xbf    = (unsigned short*)alloc((size_t)S * D * 2);
  unsigned short* wkvd_t = (unsigned short*)alloc((size_t)288 * D * 2);   // 288 x 1024
  unsigned short* wqd_t  = (unsigned short*)alloc((size_t)QC * D * 2);    // 384 x 1024
  unsigned short* wkvu_t = (unsigned short*)alloc((size_t)2048 * KVC * 2);// 2048 x 256
  unsigned short* wqu_t  = (unsigned short*)alloc((size_t)1536 * QC * 2); // 1536 x 384
  unsigned short* wo_t   = (unsigned short*)alloc((size_t)1024 * 1024 * 2);
  float* c      = (float*)alloc((size_t)S * 288 * 4);
  unsigned short* ckv_bf = (unsigned short*)alloc((size_t)S * KVC * 2);
  float* cqpre  = (float*)alloc((size_t)S * QC * 4);
  unsigned short* cq_bf  = (unsigned short*)alloc((size_t)S * QC * 2);
  float* kv     = (float*)alloc((size_t)S * 2048 * 4);
  float* qp     = (float*)alloc((size_t)S * 1536 * 4);
  unsigned short* qf  = (unsigned short*)alloc((size_t)32 * S * 64 * 2);  // [hh][s][64]
  unsigned short* kf  = (unsigned short*)alloc((size_t)32 * S * 64 * 2);  // [hh][s][64]
  unsigned short* vT  = (unsigned short*)alloc((size_t)16 * 64 * S * 2);  // [h][d][s]
  float* L      = (float*)alloc((size_t)2 * S * S * 4);
  unsigned short* A01 = (unsigned short*)alloc((size_t)2 * S * S * 2);
  unsigned short* AF  = (unsigned short*)alloc((size_t)S * S * 2);
  float* gpart  = (float*)alloc((size_t)16 * S * 4);
  float* g      = (float*)alloc((size_t)S * 4);
  float* lam    = (float*)alloc(256);
  float* attn_o = (float*)alloc((size_t)S * 1024 * 4);
  unsigned short* obf = (unsigned short*)alloc((size_t)S * 1024 * 2);

  // x -> bf16 (A operand stays row-major)
  cvt_bf16_kernel<<<(unsigned)(((long long)S * D + 255) / 256), 256, 0, stream>>>(
      x, xbf, (long long)S * D);

  // weights -> bf16, transposed to NxK for the Bt operand
  auto cvt_t = [&](const float* src, long long ldin, unsigned short* dst, int R, int Cc) {
    dim3 grid((Cc + 31) / 32, (R + 31) / 32);
    cvt_transpose_kernel<<<grid, dim3(32, 8), 0, stream>>>(src, ldin, dst, R, Cc);
  };
  cvt_t(Wkvd, 288, wkvd_t, D, 288);
  cvt_t(Wqd, QC, wqd_t, D, QC);
  cvt_t(Wkvup, 2048, wkvu_t, KVC, 2048);
  cvt_t(Wqup, 1536, wqu_t, QC, 1536);
  cvt_t(Wo, 1024, wo_t, 1024, 1024);

  // c = x @ W_kv_down  (2048 x 288 x 1024)
  launch_gemm(xbf, D, 0, wkvd_t, D, 0, c, 288, 0, S, 288, D, 0, 1, stream);
  rmsnorm_kernel<<<S, 256, 0, stream>>>(c, 288, kvw, ckv_bf, KVC, KVC);
  // c_q_pre = x @ W_q_down (2048 x 384 x 1024)
  launch_gemm(xbf, D, 0, wqd_t, D, 0, cqpre, QC, 0, S, QC, D, 0, 1, stream);
  rmsnorm_kernel<<<S, 256, 0, stream>>>(cqpre, QC, qw, cq_bf, QC, QC);
  // kv = c_kv @ W_kv_up (2048 x 2048 x 256)
  launch_gemm(ckv_bf, KVC, 0, wkvu_t, KVC, 0, kv, 2048, 0, S, 2048, KVC, 0, 1, stream);
  // qp = c_q @ W_q_up (2048 x 1536 x 384)
  launch_gemm(cq_bf, QC, 0, wqu_t, QC, 0, qp, 1536, 0, S, 1536, QC, 0, 1, stream);

  // q_f / k_f (RoPE) and v^T per head (tiled transpose out of kv)
  build_qk_kernel<<<256, 256, 0, stream>>>(qp, kv, c, freqs, qf, kf);
  for (int h = 0; h < NH; h++)
    cvt_t(kv + h * 128 + 64, 2048, vT + (long long)h * 64 * S, S, 64);
  lambda_kernel<<<1, 32, 0, stream>>>(lk1, lk2, lq1, lq2, lam);

  // attention, one head-pair at a time
  for (int p = 0; p < NH; p++) {
    int h0 = 2 * p;
    // logits for both half-heads: q_f(2048x64) @ k_f^T -> (2048x2048), Bt = k_f itself
    launch_gemm(qf + (long long)h0 * S * 64, 64, (long long)S * 64,
                kf + (long long)h0 * S * 64, 64, (long long)S * 64,
                L, S, (long long)S * S,
                S, S, 64, 1, 2, stream);
    dim3 smgrid(S, 2);
    softmax_sign_kernel<<<smgrid, 256, 0, stream>>>(L, A01, QK_SCALE);
    colmean_part_kernel<<<dim3(S / 256, 16), 256, 0, stream>>>(A01, gpart);
    colmean_final_kernel<<<S / 256, 256, 0, stream>>>(gpart, g);
    combine_kernel<<<S, 256, 0, stream>>>(A01, A01 + (long long)S * S, g, lam, AF);
    // out_head = AF @ v_head (2048 x 64 x 2048), Bt = v^T (64 x S)
    launch_gemm(AF, S, 0, vT + (long long)p * 64 * S, S, 0,
                attn_o + (long long)p * 64, 1024, 0,
                S, 64, S, 2, 1, stream);
  }

  // per-(s,h) RMSNorm over HD=64 with norm_w, then final projection
  rmsnorm_kernel<<<S * NH, 64, 0, stream>>>(attn_o, HD, normw, obf, HD, HD);
  launch_gemm(obf, 1024, 0, wo_t, 1024, 0, out, 1024, 0, S, 1024, 1024, 0, 1, stream);
}
